// IC3Net_28716151341139
// MI455X (gfx1250) — compile-verified, matrix operand order checked by reading, not run on hardware
//
#include <hip/hip_runtime.h>

#define NAGENT 2048
#define HID    2048
#define OBSZ   1024
#define ADIM   64
#define COMM_ITERS 3

typedef __attribute__((ext_vector_type(16))) __bf16          v16bf;
typedef __attribute__((ext_vector_type(8)))  float           v8f;
typedef __attribute__((ext_vector_type(8)))  unsigned short  v8us;

union Frag16 { v16bf bf; v8us us[2]; };

#define WMMA_BF16(A, B, C) \
  __builtin_amdgcn_wmma_f32_16x16x32_bf16(false, (A), false, (B), (short)0, (C), false, false)

__device__ __forceinline__ unsigned short f2bf_bits(float x) {
  unsigned u = __builtin_bit_cast(unsigned, x);
  u += 0x7FFFu + ((u >> 16) & 1u);          // round-to-nearest-even
  return (unsigned short)(u >> 16);
}

__device__ __forceinline__ float sigmoidf_(float x) {
  return 1.0f / (1.0f + __expf(-x));
}

// NT fragment load: operand stored row-major with K contiguous.
// 16-bit WMMA lane layout (ISA 7.12.2): lane L holds row (L&15); first 8
// elements are K in [kk + 8*(L>>4), +8), next 8 are K in [kk+16+8*(L>>4), +8).
__device__ __forceinline__ v16bf load_frag_nt(const unsigned short* __restrict__ base,
                                              int row0, int ld, int kk, int lane) {
  const int r  = row0 + (lane & 15);
  const int kh = kk + ((lane >> 4) << 3);
  const unsigned short* p = base + (size_t)r * (size_t)ld + kh;
  Frag16 f;
  f.us[0] = *(const v8us*)(p);
  f.us[1] = *(const v8us*)(p + 16);
  return f.bf;
}

// ---------------- conversion / setup kernels ----------------

__global__ __launch_bounds__(256) void k_f32_to_bf16(const float* __restrict__ src,
                                                     unsigned short* __restrict__ dst,
                                                     size_t n) {
  for (size_t i = (size_t)blockIdx.x * blockDim.x + threadIdx.x; i < n;
       i += (size_t)gridDim.x * blockDim.x)
    dst[i] = f2bf_bits(src[i]);
}

// wcat[g, k] : g in [0,4H), k in [0,2H);  k<H -> w_ih[g,k], else w_hh[g,k-H]
__global__ __launch_bounds__(256) void k_build_wcat(const float* __restrict__ w_ih,
                                                    const float* __restrict__ w_hh,
                                                    unsigned short* __restrict__ wcat) {
  size_t idx = (size_t)blockIdx.x * 256 + threadIdx.x;   // exact: 4H * 2H threads
  int g = (int)(idx >> 12);          // /4096
  int k = (int)(idx & 4095);
  float v = (k < HID) ? w_ih[(size_t)g * HID + k]
                      : w_hh[(size_t)g * HID + (k - HID)];
  wcat[idx] = f2bf_bits(v);
}

__global__ __launch_bounds__(256) void k_alive_scale(const float* __restrict__ alive,
                                                     float* __restrict__ scal) {
  __shared__ float red[256];
  float s = 0.f;
  for (int i = threadIdx.x; i < NAGENT; i += 256) s += alive[i];
  red[threadIdx.x] = s;
  __syncthreads();
  for (int o = 128; o > 0; o >>= 1) {
    if ((int)threadIdx.x < o) red[threadIdx.x] += red[threadIdx.x + o];
    __syncthreads();
  }
  if (threadIdx.x == 0) scal[0] = 1.0f / (red[0] - 1.0f);
}

// S[k] = sum_j alive[j] * h[j,k]   (S pre-zeroed)
__global__ __launch_bounds__(256) void k_colsum(const float* __restrict__ h,
                                                const float* __restrict__ alive,
                                                float* __restrict__ S) {
  int col = blockIdx.x * 256 + threadIdx.x;
  int r0  = blockIdx.y * 128;
  float s = 0.f;
  for (int j = 0; j < 128; ++j) {
    int row = r0 + j;
    s += alive[row] * h[(size_t)row * HID + col];
  }
  atomicAdd(&S[col], s);
}

// acat[n, 0:H) = bf16(e + comm),  acat[n, H:2H) = bf16(h)
// comm = alive[n] * (S[k] - alive[n]*h[n,k]) * inv(n_alive-1)   (gate == 1 exactly)
__global__ __launch_bounds__(256) void k_build_acat(const float* __restrict__ e,
                                                    const float* __restrict__ h,
                                                    const float* __restrict__ S,
                                                    const float* __restrict__ alive,
                                                    const float* __restrict__ scal,
                                                    unsigned short* __restrict__ acat,
                                                    int use_comm) {
  size_t idx = (size_t)blockIdx.x * 256 + threadIdx.x;   // exact: NAGENT*HID threads
  int n = (int)(idx >> 11);          // /2048
  int k = (int)(idx & 2047);
  float hv  = h[idx];
  float inp = e[idx];
  if (use_comm) {
    float ai = alive[n];
    inp += ai * (S[k] - ai * hv) * scal[0];
  }
  size_t rb = (size_t)n * (2 * HID);
  acat[rb + k]       = f2bf_bits(inp);
  acat[rb + HID + k] = f2bf_bits(hv);
}

// ---------------- WMMA GEMM kernels ----------------

// e = tanh(obs @ enc_w^T + enc_b) : M=2048, N=2048, K=1024 (NT)
// Wave computes a 2x2 grid of 16x16 tiles (32x32); block = 8 waves (2x4) = 64x128.
__global__ __launch_bounds__(256) void k_enc_gemm_tanh(const unsigned short* __restrict__ obsb,
                                                       const unsigned short* __restrict__ encwb,
                                                       const float* __restrict__ enc_b,
                                                       float* __restrict__ e) {
  const int lane = threadIdx.x & 31;
  const int w    = threadIdx.x >> 5;
  const int m0   = blockIdx.x * 64  + (w >> 2) * 32;
  const int n0   = blockIdx.y * 128 + (w & 3) * 32;
  v8f acc[2][2] = {};
  for (int kk = 0; kk < OBSZ; kk += 32) {
    v16bf a0 = load_frag_nt(obsb,  m0,      OBSZ, kk, lane);
    v16bf a1 = load_frag_nt(obsb,  m0 + 16, OBSZ, kk, lane);
    v16bf b0 = load_frag_nt(encwb, n0,      OBSZ, kk, lane);
    v16bf b1 = load_frag_nt(encwb, n0 + 16, OBSZ, kk, lane);
    acc[0][0] = WMMA_BF16(a0, b0, acc[0][0]);
    acc[0][1] = WMMA_BF16(a0, b1, acc[0][1]);
    acc[1][0] = WMMA_BF16(a1, b0, acc[1][0]);
    acc[1][1] = WMMA_BF16(a1, b1, acc[1][1]);
  }
#pragma unroll
  for (int t = 0; t < 2; ++t) {
    const int mbase = m0 + t * 16 + (lane >> 4) * 8;
#pragma unroll
    for (int u = 0; u < 2; ++u) {
      const int n = n0 + u * 16 + (lane & 15);
      const float bias = enc_b[n];
#pragma unroll
      for (int r = 0; r < 8; ++r)
        e[(size_t)(mbase + r) * HID + n] = tanhf(acc[t][u][r] + bias);
    }
  }
}

// Fused LSTM step: z = acat @ wcat^T (M=2048, N=8192 as 4 gate blocks, K=4096),
// then cell/h pointwise update in the epilogue.
// Wave computes 2 M-tiles x 4 gates; B frags reused across M, A frags across gates.
__global__ __launch_bounds__(256) void k_lstm_gemm(const unsigned short* __restrict__ acat,
                                                   const unsigned short* __restrict__ wcat,
                                                   const float* __restrict__ b_ih,
                                                   const float* __restrict__ b_hh,
                                                   float* __restrict__ cell,
                                                   float* __restrict__ h) {
  const int lane = threadIdx.x & 31;
  const int w    = threadIdx.x >> 5;
  const int m0   = blockIdx.x * 64 + (w >> 2) * 32;
  const int c0   = blockIdx.y * 64 + (w & 3) * 16;
  v8f acc[2][4] = {};
  for (int kk = 0; kk < 2 * HID; kk += 32) {
    v16bf a0 = load_frag_nt(acat, m0,      2 * HID, kk, lane);
    v16bf a1 = load_frag_nt(acat, m0 + 16, 2 * HID, kk, lane);
#pragma unroll
    for (int g = 0; g < 4; ++g) {
      v16bf b = load_frag_nt(wcat, g * HID + c0, 2 * HID, kk, lane);
      acc[0][g] = WMMA_BF16(a0, b, acc[0][g]);
      acc[1][g] = WMMA_BF16(a1, b, acc[1][g]);
    }
  }
  const int c  = c0 + (lane & 15);
  const float bi = b_ih[c]           + b_hh[c];
  const float bf = b_ih[HID + c]     + b_hh[HID + c];
  const float bg = b_ih[2 * HID + c] + b_hh[2 * HID + c];
  const float bo = b_ih[3 * HID + c] + b_hh[3 * HID + c];
#pragma unroll
  for (int t = 0; t < 2; ++t) {
    const int mbase = m0 + t * 16 + (lane >> 4) * 8;
#pragma unroll
    for (int r = 0; r < 8; ++r) {
      size_t idx = (size_t)(mbase + r) * HID + c;
      float zi = acc[t][0][r] + bi;
      float zf = acc[t][1][r] + bf;
      float zg = acc[t][2][r] + bg;
      float zo = acc[t][3][r] + bo;
      float cn = sigmoidf_(zf) * cell[idx] + sigmoidf_(zi) * tanhf(zg);
      cell[idx] = cn;
      h[idx]    = sigmoidf_(zo) * tanhf(cn);
    }
  }
}

// logits = h @ act_w^T + act_b : M=2048, N=64, K=2048 (NT)
__global__ __launch_bounds__(256) void k_act_gemm(const unsigned short* __restrict__ hb,
                                                  const unsigned short* __restrict__ actwb,
                                                  const float* __restrict__ act_b,
                                                  float* __restrict__ logits) {
  const int lane = threadIdx.x & 31;
  const int w    = threadIdx.x >> 5;
  const int m0   = blockIdx.x * 32 + (w >> 2) * 16;
  const int n0   = (w & 3) * 16;                      // N = 64 covered by 4 waves
  v8f acc = {};
  for (int kk = 0; kk < HID; kk += 32) {
    v16bf a = load_frag_nt(hb,    m0, HID, kk, lane);
    v16bf b = load_frag_nt(actwb, n0, HID, kk, lane);
    acc = WMMA_BF16(a, b, acc);
  }
  const int n     = n0 + (lane & 15);
  const int mbase = m0 + (lane >> 4) * 8;
  const float bias = act_b[n];
#pragma unroll
  for (int r = 0; r < 8; ++r)
    logits[(size_t)(mbase + r) * ADIM + n] = acc[r] + bias;
}

// ---------------- heads ----------------

__global__ __launch_bounds__(256) void k_logsoftmax(const float* __restrict__ logits,
                                                    float* __restrict__ out) {
  const int w = threadIdx.x >> 5, lane = threadIdx.x & 31;
  const int row = blockIdx.x * 8 + w;
  const float* p = logits + (size_t)row * ADIM;
  float a = p[lane], b = p[lane + 32];
  float mx = fmaxf(a, b);
  for (int o = 16; o > 0; o >>= 1) mx = fmaxf(mx, __shfl_xor(mx, o, 32));
  float se = __expf(a - mx) + __expf(b - mx);
  for (int o = 16; o > 0; o >>= 1) se += __shfl_xor(se, o, 32);
  float lse = mx + __logf(se);
  out[(size_t)row * ADIM + lane]      = a - lse;
  out[(size_t)row * ADIM + lane + 32] = b - lse;
}

__global__ __launch_bounds__(256) void k_value(const float* __restrict__ h,
                                               const float* __restrict__ val_w,
                                               const float* __restrict__ val_b,
                                               float* __restrict__ out) {
  const int w = threadIdx.x >> 5, lane = threadIdx.x & 31;
  const int row = blockIdx.x * 8 + w;
  const float* p = h + (size_t)row * HID;
  float s = 0.f;
  for (int k = lane; k < HID; k += 32) s += p[k] * val_w[k];
  for (int o = 16; o > 0; o >>= 1) s += __shfl_xor(s, o, 32);
  if (lane == 0) out[row] = s + val_b[0];
}

// ---------------- launcher ----------------

extern "C" void kernel_launch(void* const* d_in, const int* in_sizes, int n_in,
                              void* d_out, int out_size, void* d_ws, size_t ws_size,
                              hipStream_t stream) {
  (void)in_sizes; (void)n_in; (void)out_size; (void)ws_size;
  const float* obs   = (const float*)d_in[0];
  const float* alive = (const float*)d_in[1];
  const float* enc_w = (const float*)d_in[2];
  const float* enc_b = (const float*)d_in[3];
  // d_in[4] = g_w, d_in[5] = g_b are provably dead: gate = ceil(sigmoid(x)) == 1
  // for all reachable x (|x| << 104, the fp32 sigmoid-underflow threshold).
  const float* w_ih  = (const float*)d_in[6];
  const float* w_hh  = (const float*)d_in[7];
  const float* b_ih  = (const float*)d_in[8];
  const float* b_hh  = (const float*)d_in[9];
  const float* act_w = (const float*)d_in[10];
  const float* act_b = (const float*)d_in[11];
  const float* val_w = (const float*)d_in[12];
  const float* val_b = (const float*)d_in[13];

  float* out_logp = (float*)d_out;
  float* out_val  = out_logp + (size_t)NAGENT * ADIM;

  char* ws = (char*)d_ws;
  size_t off = 0;
  auto take = [&](size_t bytes) -> char* {
    char* p = ws + off;
    off = (off + bytes + 255) & ~(size_t)255;
    return p;
  };

  unsigned short* wcatb = (unsigned short*)take((size_t)4 * HID * 2 * HID * 2); // 64 MB
  unsigned short* encwb = (unsigned short*)take((size_t)HID * OBSZ * 2);
  unsigned short* actwb = (unsigned short*)take((size_t)ADIM * HID * 2);
  unsigned short* obsb  = (unsigned short*)take((size_t)NAGENT * OBSZ * 2);
  unsigned short* acatb = (unsigned short*)take((size_t)NAGENT * 2 * HID * 2);
  unsigned short* hb    = (unsigned short*)take((size_t)NAGENT * HID * 2);
  float* e      = (float*)take((size_t)NAGENT * HID * 4);
  float* h      = (float*)take((size_t)NAGENT * HID * 4);
  float* cell   = (float*)take((size_t)NAGENT * HID * 4);
  float* S      = (float*)take((size_t)HID * 4);
  float* scal   = (float*)take(256);
  float* logits = (float*)take((size_t)NAGENT * ADIM * 4);

  const dim3 blk(256);

  // One-time (per call) weight/input precision conversion; weights then live in L2.
  k_build_wcat<<<dim3((4 * HID * (2 * HID)) / 256), blk, 0, stream>>>(w_ih, w_hh, wcatb);
  k_f32_to_bf16<<<dim3(8192), blk, 0, stream>>>(enc_w, encwb, (size_t)HID * OBSZ);
  k_f32_to_bf16<<<dim3(8192), blk, 0, stream>>>(obs, obsb, (size_t)NAGENT * OBSZ);
  k_f32_to_bf16<<<dim3(512),  blk, 0, stream>>>(act_w, actwb, (size_t)ADIM * HID);
  k_alive_scale<<<1, blk, 0, stream>>>(alive, scal);
  hipMemsetAsync(h,    0, (size_t)NAGENT * HID * 4, stream);
  hipMemsetAsync(cell, 0, (size_t)NAGENT * HID * 4, stream);

  k_enc_gemm_tanh<<<dim3(NAGENT / 64, HID / 128), blk, 0, stream>>>(obsb, encwb, enc_b, e);

  for (int it = 0; it < COMM_ITERS; ++it) {
    const int use_comm = (it > 0);
    if (use_comm) {
      hipMemsetAsync(S, 0, HID * 4, stream);
      k_colsum<<<dim3(HID / 256, NAGENT / 128), blk, 0, stream>>>(h, alive, S);
    }
    k_build_acat<<<dim3((NAGENT * HID) / 256), blk, 0, stream>>>(e, h, S, alive, scal,
                                                                 acatb, use_comm);
    k_lstm_gemm<<<dim3(NAGENT / 64, HID / 64), blk, 0, stream>>>(acatb, wcatb,
                                                                 b_ih, b_hh, cell, h);
  }

  k_f32_to_bf16<<<dim3(16384), blk, 0, stream>>>(h, hb, (size_t)NAGENT * HID);
  k_act_gemm<<<dim3(NAGENT / 32, 1), blk, 0, stream>>>(hb, actwb, act_b, logits);
  k_logsoftmax<<<dim3(NAGENT / 8), blk, 0, stream>>>(logits, out_logp);
  k_value<<<dim3(NAGENT / 8), blk, 0, stream>>>(h, val_w, val_b, out_val);
}
